// Net_58660663329250
// MI455X (gfx1250) — matrix operand order repaired and checked by reading, not compiled, not run on hardware
//
#include <hip/hip_runtime.h>
#include <math.h>

typedef __attribute__((ext_vector_type(16))) _Float16 v16h;
typedef __attribute__((ext_vector_type(8)))  _Float16 v8h;
typedef __attribute__((ext_vector_type(8)))  float    v8f;

#define KS 5

__device__ __forceinline__ float eluf(float x) { return x > 0.f ? x : expf(x) - 1.f; }

// ---------------------------------------------------------------- fill
__global__ void fill_kernel(float* p, float v, long n) {
    long i = (long)blockIdx.x * blockDim.x + threadIdx.x;
    long stride = (long)gridDim.x * blockDim.x;
    for (; i < n; i += stride) p[i] = v;
}

// ------------------------------------------------ f16 operand pack kernels
// Ah[Np, Kp] = zero-padded f16 copy of A[N, K]
__global__ void pack_a_f16(const float* __restrict__ A, _Float16* __restrict__ Ah,
                           int N, int K, int Np, int Kp)
{
    long i = (long)blockIdx.x * blockDim.x + threadIdx.x;
    long tot = (long)Np * Kp;
    long stride = (long)gridDim.x * blockDim.x;
    for (; i < tot; i += stride) {
        int r = (int)(i / Kp), k = (int)(i % Kp);
        Ah[i] = (r < N && k < K) ? (_Float16)A[(size_t)r * K + k] : (_Float16)0.f;
    }
}
// Bt[Cop, Kp] = zero-padded f16 TRANSPOSE of B[K, Co]  (column-major for GEMM)
__global__ void pack_bt_f16(const float* __restrict__ B, _Float16* __restrict__ Bt,
                            int K, int Co, int Cop, int Kp)
{
    long i = (long)blockIdx.x * blockDim.x + threadIdx.x;
    long tot = (long)Cop * Kp;
    long stride = (long)gridDim.x * blockDim.x;
    for (; i < tot; i += stride) {
        int c = (int)(i / Kp), k = (int)(i % Kp);
        Bt[i] = (k < K && c < Co) ? (_Float16)B[(size_t)k * Co + c] : (_Float16)0.f;
    }
}

// ------------------------------------------- spline edge messages (scatter)
// One wave32 per edge; lanes stride output channels. src/dst forced wave-
// uniform via readfirstlane so feature-row loads can use the scalar path.
__global__ void spline_edge_msg(const float* __restrict__ x,
                                const int* __restrict__ src,
                                const int* __restrict__ dst,
                                const float* __restrict__ pseudo,
                                const float* __restrict__ W,   // [125, Cin, Co]
                                float* __restrict__ agg,       // [N, Co]
                                float* __restrict__ deg,       // [N]
                                int E, int Cin, int Co)
{
    const int wave = threadIdx.x >> 5;
    const int lane = threadIdx.x & 31;
    const int e = blockIdx.x * (blockDim.x >> 5) + wave;
    if (e >= E) return;

    // prefetch next tile's source feature row (global_prefetch_b8)
    if (e + 8 < E) __builtin_prefetch(x + (size_t)src[e + 8] * Cin, 0, 1);

    float bs[8]; int wk[8];
    {
        float f[3]; int i0[3];
#pragma unroll
        for (int d = 0; d < 3; ++d) {
            float t  = pseudo[(size_t)e * 3 + d] * (float)(KS - 1);
            float fl = floorf(t);
            fl = fminf(fmaxf(fl, 0.f), (float)(KS - 2));
            i0[d] = (int)fl; f[d] = t - fl;
        }
#pragma unroll
        for (int c = 0; c < 8; ++c) {
            float b = 1.f; int k = 0; int p5 = 1;
#pragma unroll
            for (int d = 0; d < 3; ++d) {
                int bit = (c >> d) & 1;
                b *= bit ? f[d] : (1.f - f[d]);
                k += (i0[d] + bit) * p5;
                p5 *= KS;
            }
            bs[c] = b; wk[c] = k;
        }
    }
    const int sN = __builtin_amdgcn_readfirstlane(src[e]);
    const int dN = __builtin_amdgcn_readfirstlane(dst[e]);
    const float* xrow = x + (size_t)sN * Cin;

    for (int o = lane; o < Co; o += 32) {
        float m = 0.f;
#pragma unroll
        for (int s = 0; s < 8; ++s) {
            const float* Wk = W + ((size_t)wk[s] * Cin) * Co + o;
            float dotv = 0.f;
            for (int c = 0; c < Cin; ++c) dotv += xrow[c] * Wk[(size_t)c * Co];
            m += bs[s] * dotv;
        }
        atomicAdd(&agg[(size_t)dN * Co + o], m);
    }
    if (lane == 0) atomicAdd(&deg[dN], 1.f);
}

// --------------------------------- WMMA GEMM with fused spline-conv epilogue
// out = act( agg/deg + A @ B + bias + res ), any term optional.
// Operands pre-packed: Ah[Np,Kp] f16 row-major, Bt[Cop,Kp] f16 (B transposed).
// K-loop body = 4x global_load_b128 + v_wmma_f32_16x16x32_f16, no predication.
// Fragment layouts (CDNA5 16-bit):
//   A 16x32: lane<16 holds M=lane, K = hiA+{0..7} and hiA+{16..23}, hiA=8*(l>=16)
//   B 32x16: lane holds N=lane%16, K = hiB+{0..7} and hiB+{16..23}, hiB=8*(l>=16)
__global__ void wmma_gemm_ep(const _Float16* __restrict__ Ah,  // [Np, Kp]
                             const _Float16* __restrict__ Bt,  // [Cop, Kp]
                             const float* __restrict__ bias,   // [Co] or null
                             const float* __restrict__ agg,    // [N, Co] or null
                             const float* __restrict__ deg,    // [N] or null
                             const float* __restrict__ res,    // [N, Co] or null
                             float* __restrict__ out,          // [N, Co]
                             int N, int Co, int Kp, int act)
{
    const int tn_cnt = (Co + 15) >> 4;
    const int tile = blockIdx.x;
    const int tm = (tile / tn_cnt) << 4;
    const int tn = (tile % tn_cnt) << 4;
    const int l = threadIdx.x;               // 0..31 (wave32)

    const int hiA = (l >= 16) ? 8 : 0;
    const int hiB = (l >= 16) ? 8 : 0;
    const _Float16* ap = Ah + (size_t)(tm + (l & 15)) * Kp + hiA;
    const _Float16* bp = Bt + (size_t)(tn + (l & 15)) * Kp + hiB;

    union V16 { v16h v; v8h h[2]; };
    v8f acc = {};
    for (int k0 = 0; k0 < Kp; k0 += 32) {
        V16 a, b;
        a.h[0] = *(const v8h*)(ap + k0);        // K = hiA + 0..7
        a.h[1] = *(const v8h*)(ap + k0 + 16);   // K = hiA + 16..23
        b.h[0] = *(const v8h*)(bp + k0);        // K = hiB + 0..7
        b.h[1] = *(const v8h*)(bp + k0 + 16);   // K = hiB + 16..23
        acc = __builtin_amdgcn_wmma_f32_16x16x32_f16(
            /*neg_a=*/false, a.v, /*neg_b=*/false, b.v,
            /*c_mod=*/(short)0, acc, /*reuse_a=*/false, /*reuse_b=*/false);
    }

#pragma unroll
    for (int r = 0; r < 8; ++r) {
        int m = tm + ((l >= 16) ? 8 : 0) + r;
        int n = tn + (l & 15);
        if (m < N && n < Co) {
            float v = acc[r];
            if (agg)  v += agg[(size_t)m * Co + n] / fmaxf(deg[m], 1.f);
            if (bias) v += bias[n];
            if (res)  v += res[(size_t)m * Co + n];
            if (act)  v = eluf(v);
            out[(size_t)m * Co + n] = v;
        }
    }
}

// ---------------------------------------------------------------- pooling
__device__ __forceinline__ void atomicMaxF(float* addr, float v) {
    if (v >= 0.f) atomicMax((int*)addr, __float_as_int(v));
    else          atomicMin((unsigned int*)addr, __float_as_uint(v));
}

__global__ void pool_scatter_x(const float* __restrict__ x, const int* __restrict__ cluster,
                               float* __restrict__ xc, int Nf, int C)
{
    long i = (long)blockIdx.x * blockDim.x + threadIdx.x;
    long tot = (long)Nf * C;
    if (i >= tot) return;
    int nf = (int)(i / C), ch = (int)(i % C);
    atomicMaxF(&xc[(size_t)cluster[nf] * C + ch], x[i]);
}

__global__ void pool_scatter_pos(const float* __restrict__ pos, const int* __restrict__ cluster,
                                 float* __restrict__ possum, float* __restrict__ cnt, int Nf)
{
    int i = blockIdx.x * blockDim.x + threadIdx.x;
    if (i >= Nf) return;
    int c = cluster[i];
    atomicAdd(&possum[(size_t)c * 3 + 0], pos[(size_t)i * 3 + 0]);
    atomicAdd(&possum[(size_t)c * 3 + 1], pos[(size_t)i * 3 + 1]);
    atomicAdd(&possum[(size_t)c * 3 + 2], pos[(size_t)i * 3 + 2]);
    atomicAdd(&cnt[c], 1.f);
}

__global__ void pool_pos_final(float* posc, const float* possum, const float* cnt, int Nc)
{
    int i = blockIdx.x * blockDim.x + threadIdx.x;
    if (i >= Nc) return;
    float c = cnt[i];
    posc[(size_t)i * 3 + 0] = possum[(size_t)i * 3 + 0] / c;
    posc[(size_t)i * 3 + 1] = possum[(size_t)i * 3 + 1] / c;
    posc[(size_t)i * 3 + 2] = possum[(size_t)i * 3 + 2] / c;
}

// ---------------------------------------------------------------- concat
__global__ void concat_ones_k(const float* __restrict__ in, float* __restrict__ out, int N, int C)
{
    long i = (long)blockIdx.x * blockDim.x + threadIdx.x;
    long tot = (long)N * (C + 1);
    if (i >= tot) return;
    int n = (int)(i / (C + 1)), c = (int)(i % (C + 1));
    out[i] = (c < C) ? in[(size_t)n * C + c] : 1.f;
}

__global__ void concat_gather_k(const float* __restrict__ a, const int* __restrict__ idx, int Ca,
                                const float* __restrict__ b, int Cb,
                                float* __restrict__ out, int N)
{
    long i = (long)blockIdx.x * blockDim.x + threadIdx.x;
    int Ct = Ca + Cb;
    long tot = (long)N * Ct;
    if (i >= tot) return;
    int n = (int)(i / Ct), c = (int)(i % Ct);
    out[i] = (c < Ca) ? a[(size_t)idx[n] * Ca + c] : b[(size_t)n * Cb + (c - Ca)];
}

// ------------------------------------------------- head: log_softmax + split
__global__ void head_kernel(const float* __restrict__ o18, float* __restrict__ cls,
                            float* __restrict__ bb, int N)
{
    int n = blockIdx.x * blockDim.x + threadIdx.x;
    if (n >= N) return;
    const float* r = o18 + (size_t)n * 18;
#pragma unroll
    for (int a = 0; a < 2; ++a) {
        float p0 = r[2 * a], p1 = r[2 * a + 1];
        float mx = fmaxf(p0, p1);
        float ls = mx + logf(expf(p0 - mx) + expf(p1 - mx));
        cls[(size_t)(2 * n + a) * 2 + 0] = p0 - ls;
        cls[(size_t)(2 * n + a) * 2 + 1] = p1 - ls;
    }
#pragma unroll
    for (int a = 0; a < 2; ++a)
#pragma unroll
        for (int j = 0; j < 7; ++j)
            bb[(size_t)(2 * n + a) * 7 + j] = r[4 + 7 * a + j];
}

__global__ void copy_kernel(const float* __restrict__ in, float* __restrict__ out, long n)
{
    long i = (long)blockIdx.x * blockDim.x + threadIdx.x;
    if (i < n) out[i] = in[i];
}

// ================================================================ driver
extern "C" void kernel_launch(void* const* d_in, const int* in_sizes, int n_in,
                              void* d_out, int out_size, void* d_ws, size_t ws_size,
                              hipStream_t stream)
{
    const int N0 = 50000, N1 = 12000, N2 = 4000, N3 = 1200, N4 = 400;

    const float* x_in = (const float*)d_in[0];
    const float* pos0 = (const float*)d_in[1];
    const int*   srcp[5]; const int* dstp[5]; const float* ps[5]; int E[5];
    for (int l = 0; l < 5; ++l) {
        srcp[l] = (const int*)d_in[2 + 3 * l];
        dstp[l] = (const int*)d_in[3 + 3 * l];
        ps[l]   = (const float*)d_in[4 + 3 * l];
        E[l]    = in_sizes[2 + 3 * l];
    }
    const int* clus[4];
    for (int l = 0; l < 4; ++l) clus[l] = (const int*)d_in[17 + l];

    struct Conv { const float *W, *root, *bias; };
    Conv cv[12];   // conv1,conv2,conv22,conv3,conv32,conv4,conv42,conv5,conv52,rpn1,rpn2,rpn4
    int pi = 21;
    for (int i = 0; i < 12; ++i) { cv[i] = { (const float*)d_in[pi], (const float*)d_in[pi + 1], (const float*)d_in[pi + 2] }; pi += 3; }
    const float* fc2w = (const float*)d_in[pi++]; const float* fc2b = (const float*)d_in[pi++];
    const float* fc3w = (const float*)d_in[pi++]; const float* fc3b = (const float*)d_in[pi++];
    const float* sk1w = (const float*)d_in[pi++]; const float* sk1b = (const float*)d_in[pi++];
    const float* sk2w = (const float*)d_in[pi++]; const float* sk2b = (const float*)d_in[pi++];

    // ---- workspace bump allocator (64B-aligned blocks)
    size_t off = 0;
    auto allocB = [&](size_t bytes) -> void* {
        void* p = (void*)((char*)d_ws + off);
        off += (bytes + 63) & ~(size_t)63;
        return p;
    };
    auto alloc  = [&](size_t n) -> float*     { return (float*)allocB(n * sizeof(float)); };
    auto allocH = [&](size_t n) -> _Float16*  { return (_Float16*)allocB(n * sizeof(_Float16)); };

    float* agg   = alloc((size_t)N0 * 32);   // largest agg need (conv1)
    float* deg   = alloc(N0);
    float* cnt   = alloc(N1);
    _Float16* Ah = allocH((size_t)N0 * 32);  // largest packed A (conv1 root, Kp=32)
    _Float16* Bt = allocH(32768);            // largest packed B^T (rpn1: 64x160)
    float* x0    = alloc((size_t)N0 * 32);
    float* x1p   = alloc((size_t)N1 * 32);  float* pos1 = alloc((size_t)N1 * 3);
    float* xcat1 = alloc((size_t)N1 * 33);
    float* t1    = alloc((size_t)N1 * 64);
    float* t2    = alloc((size_t)N1 * 64);
    float* x1    = alloc((size_t)N1 * 64);
    float* x2p   = alloc((size_t)N2 * 64);  float* pos2 = alloc((size_t)N2 * 3);
    float* xcat2 = alloc((size_t)N2 * 65);
    float* t3    = alloc((size_t)N2 * 64);
    float* x2    = alloc((size_t)N2 * 64);
    float* x3p   = alloc((size_t)N3 * 64);  float* pos3 = alloc((size_t)N3 * 3);
    float* xcat3 = alloc((size_t)N3 * 65);
    float* t4    = alloc((size_t)N3 * 64);
    float* x3    = alloc((size_t)N3 * 64);
    float* x4p   = alloc((size_t)N4 * 64);  float* pos4 = alloc((size_t)N4 * 3);
    float* xcat4 = alloc((size_t)N4 * 65);
    float* t5    = alloc((size_t)N4 * 96);
    float* t6    = alloc((size_t)N4 * 96);
    float* x4    = alloc((size_t)N4 * 96);
    float* fc3x3 = alloc((size_t)N3 * 64);
    float* u3    = alloc((size_t)N3 * 160);
    float* x3r   = alloc((size_t)N3 * 64);
    float* fc2x2 = alloc((size_t)N2 * 64);
    float* u2    = alloc((size_t)N2 * 128);
    float* x2r   = alloc((size_t)N2 * 64);
    float* o18   = alloc((size_t)N2 * 18);
    (void)ws_size; (void)n_in; (void)out_size;

    auto fill = [&](float* p, float v, long n) {
        int blocks = (int)((n + 255) / 256);
        fill_kernel<<<blocks, 256, 0, stream>>>(p, v, n);
    };
    // pack f32 operands into padded f16 Ah/Bt, then fused WMMA GEMM + epilogue
    auto gemm = [&](const float* A, const float* B, const float* bias,
                    const float* aggp, const float* degp, const float* res,
                    float* outp, int Nn, int Cin, int Co, int act) {
        int Kp  = ((Cin + 31) / 32) * 32;
        int Np  = ((Nn + 15) / 16) * 16;
        int Cop = ((Co + 15) / 16) * 16;
        long na = (long)Np * Kp, nb = (long)Cop * Kp;
        pack_a_f16 <<<(int)((na + 255) / 256), 256, 0, stream>>>(A, Ah, Nn, Cin, Np, Kp);
        pack_bt_f16<<<(int)((nb + 255) / 256), 256, 0, stream>>>(B, Bt, Cin, Co, Cop, Kp);
        int tiles = (Np / 16) * (Cop / 16);
        wmma_gemm_ep<<<tiles, 32, 0, stream>>>(Ah, Bt, bias, aggp, degp, res, outp,
                                               Nn, Co, Kp, act);
    };
    // full spline_conv: zero agg/deg -> edge scatter -> fused WMMA root-GEMM
    auto spline = [&](const float* X, int lvl, const Conv& c, int Cin, int Co, int Nn,
                      const float* res, float* outp, int act) {
        fill(agg, 0.f, (long)Nn * Co);
        fill(deg, 0.f, Nn);
        int eb = (E[lvl] + 7) / 8;
        spline_edge_msg<<<eb, 256, 0, stream>>>(X, srcp[lvl], dstp[lvl], ps[lvl],
                                                c.W, agg, deg, E[lvl], Cin, Co);
        gemm(X, c.root, c.bias, agg, deg, res, outp, Nn, Cin, Co, act);
    };
    auto linK = [&](const float* X, const float* w, const float* b, const float* res,
                    float* outp, int Nn, int Ci, int Co, int act) {
        gemm(X, w, b, nullptr, nullptr, res, outp, Nn, Ci, Co, act);
    };
    auto pool = [&](const float* X, const float* POS, const int* cl, int Nf, int Nc, int C,
                    float* xc, float* posc) {
        fill(xc, -3.402823466e38f, (long)Nc * C);
        fill(posc, 0.f, (long)Nc * 3);
        fill(cnt, 0.f, Nc);
        pool_scatter_x<<<(int)(((long)Nf * C + 255) / 256), 256, 0, stream>>>(X, cl, xc, Nf, C);
        pool_scatter_pos<<<(Nf + 255) / 256, 256, 0, stream>>>(POS, cl, posc, cnt, Nf);
        pool_pos_final<<<(Nc + 255) / 256, 256, 0, stream>>>(posc, posc, cnt, Nc);
    };
    auto cones = [&](const float* in, float* outp, int N, int C) {
        concat_ones_k<<<(int)(((long)N * (C + 1) + 255) / 256), 256, 0, stream>>>(in, outp, N, C);
    };

    // ----------------- down path
    spline(x_in, 0, cv[0], 1, 32, N0, nullptr, x0, 1);                 // conv1 + elu
    pool(x0, pos0, clus[0], N0, N1, 32, x1p, pos1);
    cones(x1p, xcat1, N1, 32);
    spline(xcat1, 1, cv[1], 33, 64, N1, nullptr, t1, 1);               // conv2 + elu
    spline(t1,    1, cv[2], 64, 64, N1, nullptr, t2, 0);               // conv22
    linK(x1p, sk1w, sk1b, t2, x1, N1, 32, 64, 1);                      // x1 = elu(t2 + skip1)

    pool(x1, pos1, clus[1], N1, N2, 64, x2p, pos2);
    cones(x2p, xcat2, N2, 64);
    spline(xcat2, 2, cv[3], 65, 64, N2, nullptr, t3, 1);               // conv3 + elu
    spline(t3,    2, cv[4], 64, 64, N2, x2p,     x2, 1);               // x2 = elu(conv32 + x2p)

    pool(x2, pos2, clus[2], N2, N3, 64, x3p, pos3);
    cones(x3p, xcat3, N3, 64);
    spline(xcat3, 3, cv[5], 65, 64, N3, nullptr, t4, 1);               // conv4 + elu
    spline(t4,    3, cv[6], 64, 64, N3, x3p,     x3, 1);               // x3 = elu(conv42 + x3p)

    pool(x3, pos3, clus[3], N3, N4, 64, x4p, pos4);
    cones(x4p, xcat4, N4, 64);
    spline(xcat4, 4, cv[7], 65, 96, N4, nullptr, t5, 1);               // conv5 + elu
    spline(t5,    4, cv[8], 96, 96, N4, nullptr, t6, 0);               // conv52
    linK(x4p, sk2w, sk2b, t6, x4, N4, 64, 96, 1);                      // x4 = elu(t6 + skip2)

    // ----------------- up path
    linK(x3, fc3w, fc3b, nullptr, fc3x3, N3, 64, 64, 0);
    concat_gather_k<<<(int)(((long)N3 * 160 + 255) / 256), 256, 0, stream>>>(
        x4, clus[3], 96, fc3x3, 64, u3, N3);
    spline(u3, 3, cv[9], 160, 64, N3, nullptr, x3r, 1);                // rpn1 + elu

    linK(x2, fc2w, fc2b, nullptr, fc2x2, N2, 64, 64, 0);
    concat_gather_k<<<(int)(((long)N2 * 128 + 255) / 256), 256, 0, stream>>>(
        x3r, clus[2], 64, fc2x2, 64, u2, N2);
    spline(u2, 2, cv[10], 128, 64, N2, nullptr, x2r, 1);               // rpn2 + elu

    spline(x2r, 2, cv[11], 64, 18, N2, nullptr, o18, 0);               // rpn4

    // ----------------- outputs: cls(16000) | bb(56000) | pos2(12000)
    float* outF = (float*)d_out;
    head_kernel<<<(N2 + 255) / 256, 256, 0, stream>>>(o18, outF, outF + 16000, N2);
    copy_kernel<<<(int)((12000 + 255) / 256), 256, 0, stream>>>(pos2, outF + 16000 + 56000, 12000);
}
